// Seq2Seq_24240795418792
// MI455X (gfx1250) — compile-verified, matrix operand order checked by reading, not run on hardware
//
#include <hip/hip_runtime.h>
#include <hip/hip_bf16.h>

// Problem constants (from reference): V,E,H,B,S,T
#define V_  32000
#define E_  512
#define H_  1024
#define B_  16
#define S_  128
#define T_  64
#define G4  (4*H_)          // 4096 gate width
#define TD  (T_-1)          // 63 decoder steps
#define M_FC (B_*TD)        // 1008 decoder rows
#define M_FC_PAD 1024       // padded rows so TDM/LDS staging never OOBs

typedef __attribute__((ext_vector_type(16))) __bf16 bf16x16;
typedef __attribute__((ext_vector_type(8)))  float  f32x8;
typedef __attribute__((ext_vector_type(4)))  unsigned int u32x4;
typedef __attribute__((ext_vector_type(8)))  unsigned int u32x8;

// ---------------------------------------------------------------------------
// WMMA helper: V_WMMA_F32_16X16X32_BF16, D = A(16x32) * B(32x16) + C(16x16)
// ---------------------------------------------------------------------------
__device__ __forceinline__ f32x8 wmma_bf16(bf16x16 a, bf16x16 b, f32x8 c) {
  return __builtin_amdgcn_wmma_f32_16x16x32_bf16(
      /*neg_a=*/false, a, /*neg_b=*/false, b,
      /*c_mod=*/(short)0, c, /*reuse_a=*/false, /*reuse_b=*/false);
}

// A fragment (16x32 bf16, row-major source, ld = K stride in elements).
// Lane layout (ISA 7.12.2): lane<16 -> M=lane, K = k0+{0..7,16..23};
//                           lane>=16 -> M=lane-16, K = k0+{8..15,24..31}.
// B fragment (32x16) is the mirror with N in place of M, so the same loader
// works on W stored row-major [N_total, K_total] (and on LDS tiles).
__device__ __forceinline__ bf16x16 frag_ld_bf16(const __bf16* __restrict__ p,
                                                int ld, int r0, int k0, int lane) {
  const int rr = lane & 15, hf = lane >> 4;
  const __bf16* q = p + (size_t)(r0 + rr) * ld + k0 + hf * 8;
  bf16x16 v;
#pragma unroll
  for (int j = 0; j < 8; ++j) v[j] = q[j];
#pragma unroll
  for (int j = 0; j < 8; ++j) v[8 + j] = q[16 + j];
  return v;
}

__device__ __forceinline__ float sigmoidf_(float x) {
  return 1.0f / (1.0f + __expf(-x));
}

// ---------------------------------------------------------------------------
// Tensor Data Mover: DMA a [rows x 32] bf16 tile (row stride = ldk elements)
// from global memory into LDS at byte offset lds_off.  D# built per ISA ch.8;
// inline asm (VIMAGE 2-group form) so it works on both toolchains' arities.
// All inputs must be wave-uniform (SGPR-resident).
// ---------------------------------------------------------------------------
__device__ __forceinline__ void tdm_load_tile_bf16(const __bf16* gsrc,
                                                   unsigned lds_off,
                                                   unsigned rows,
                                                   unsigned ldk) {
  unsigned long long ga = (unsigned long long)(uintptr_t)gsrc;
  u32x4 g0;
  g0[0] = 1u;                                   // count=1, user mode
  g0[1] = lds_off;                              // lds_addr (bytes)
  g0[2] = (unsigned)ga;                         // global_addr[31:0]
  g0[3] = (unsigned)((ga >> 32) & 0x01FFFFFFu)  // global_addr[56:32]
        | (2u << 30);                           // type=2 ("image")
  u32x8 g1;
  g1[0] = 1u << 16;                             // data_size=1 (2 bytes)
  g1[1] = 32u << 16;                            // tensor_dim0=32 (bits 79:48 lo)
  g1[2] = rows << 16;                           // tensor_dim1 lo16 (bits 111:80)
  g1[3] = 32u << 16;                            // tile_dim0=32 (bits 127:112)
  g1[4] = rows;                                 // tile_dim1 (tile_dim2=0)
  g1[5] = ldk;                                  // tensor_dim0_stride lo32
  g1[6] = 0u;                                   // stride hi / dim1_stride lo
  g1[7] = 0u;
  asm volatile("tensor_load_to_lds %0, %1" :: "s"(g0), "s"(g1) : "memory");
}

// ---------------------------------------------------------------------------
// Small utility kernels
// ---------------------------------------------------------------------------
__global__ void k_cast_bf16(const float* __restrict__ src, __bf16* __restrict__ dst, int n) {
  int i = blockIdx.x * blockDim.x + threadIdx.x;
  if (i < n) dst[i] = (__bf16)src[i];
}

__global__ void k_bias_combine(const float* __restrict__ a, const float* __restrict__ b,
                               float* __restrict__ out, int n) {
  int i = blockIdx.x * blockDim.x + threadIdx.x;
  if (i < n) out[i] = a[i] + b[i];
}

// Embedding gather + cast: dst row r = b*outT + t, token = tok[b*srcT + t].
__global__ void k_embed(const int* __restrict__ tok, const float* __restrict__ emb,
                        __bf16* __restrict__ dst, int nrows, int outT, int srcT) {
  int idx = blockIdx.x * blockDim.x + threadIdx.x;
  if (idx >= nrows * E_) return;
  int r = idx / E_, e = idx - r * E_;
  int b = r / outT, t = r - b * outT;
  int token = tok[b * srcT + t];
  dst[(size_t)r * E_ + e] = (__bf16)emb[(size_t)token * E_ + e];
}

__global__ void k_zero_state(__bf16* __restrict__ h, float* __restrict__ c) {
  int i = blockIdx.x * blockDim.x + threadIdx.x;
  if (i < B_ * H_) { h[i] = (__bf16)0.0f; c[i] = 0.0f; }
}

// ---------------------------------------------------------------------------
// XW = X[M,K] @ Wih^T (+ bih + bhh).  Wih row-major [4096, K].
// One wave -> 16x32 output (two N-tiles sharing the A fragment).
// ---------------------------------------------------------------------------
__global__ void __launch_bounds__(256)
k_gemm_xw(const __bf16* __restrict__ A, const __bf16* __restrict__ W,
          const float* __restrict__ bias, float* __restrict__ C, int M, int K) {
  const int lane = threadIdx.x & 31;
  const int wave = blockIdx.x * (blockDim.x >> 5) + (threadIdx.x >> 5);
  const int npairs = G4 / 32;                  // 128
  const int mt = wave / npairs;
  const int np = wave - mt * npairs;
  const int m0 = mt * 16, n0 = np * 32;
  if (m0 >= M) return;

  f32x8 acc0 = {}, acc1 = {};
  for (int k0 = 0; k0 < K; k0 += 32) {
    if (k0 + 32 < K) {
      __builtin_prefetch(A + (size_t)m0 * K + k0 + 32, 0, 3);
      __builtin_prefetch(W + (size_t)n0 * K + k0 + 32, 0, 3);
    }
    bf16x16 a  = frag_ld_bf16(A, K, m0, k0, lane);
    bf16x16 b0 = frag_ld_bf16(W, K, n0, k0, lane);
    bf16x16 b1 = frag_ld_bf16(W, K, n0 + 16, k0, lane);
    acc0 = wmma_bf16(a, b0, acc0);
    acc1 = wmma_bf16(a, b1, acc1);
  }
  const int nidx = lane & 15, hf = lane >> 4;
#pragma unroll
  for (int r = 0; r < 8; ++r) {
    int m = hf * 8 + r;                        // C/D layout: VGPR r -> M = r + 8*half
    float* row = C + (size_t)(m0 + m) * G4;
    row[n0 + nidx]      = acc0[r] + bias[n0 + nidx];
    row[n0 + 16 + nidx] = acc1[r] + bias[n0 + 16 + nidx];
  }
}

// ---------------------------------------------------------------------------
// One LSTM timestep: gates[16,4096] = XW[:,t,:] + h_in @ Whh^T, then the cell
// update.  64 waves; wave w owns gate columns [16w,16w+16) for all 4 gates
// (i,f,g,o at offsets 0/H/2H/3H), so the nonlinearity happens in-register.
// Whh (bf16, 8 MB) stays resident in the 192 MB L2 across all 191 steps.
// h is ping-ponged across kernel launches (stream order = time order).
// ---------------------------------------------------------------------------
__global__ void __launch_bounds__(256)
k_lstm_step(const __bf16* __restrict__ h_in, __bf16* __restrict__ h_out,
            float* __restrict__ c, const __bf16* __restrict__ Whh,
            const float* __restrict__ XW, int steps, int t,
            __bf16* __restrict__ hs) {
  const int lane = threadIdx.x & 31;
  const int wave = blockIdx.x * (blockDim.x >> 5) + (threadIdx.x >> 5);  // 0..63
  const int j0 = wave * 16;

  f32x8 ai = {}, af = {}, ag = {}, ao = {};
  for (int k0 = 0; k0 < H_; k0 += 32) {
    if (k0 + 32 < H_) {
      __builtin_prefetch(Whh + (size_t)j0 * H_ + k0 + 32, 0, 3);
    }
    bf16x16 a  = frag_ld_bf16(h_in, H_, 0, k0, lane);
    bf16x16 bi = frag_ld_bf16(Whh, H_, j0,          k0, lane);
    bf16x16 bf = frag_ld_bf16(Whh, H_, H_ + j0,     k0, lane);
    bf16x16 bg = frag_ld_bf16(Whh, H_, 2 * H_ + j0, k0, lane);
    bf16x16 bo = frag_ld_bf16(Whh, H_, 3 * H_ + j0, k0, lane);
    ai = wmma_bf16(a, bi, ai);
    af = wmma_bf16(a, bf, af);
    ag = wmma_bf16(a, bg, ag);
    ao = wmma_bf16(a, bo, ao);
  }
  const int nidx = lane & 15, hf = lane >> 4;
  const int col = j0 + nidx;
#pragma unroll
  for (int r = 0; r < 8; ++r) {
    const int m = hf * 8 + r;                       // batch row
    const float* xw = XW + (size_t)(m * steps + t) * G4;
    float iv = ai[r] + xw[col];
    float fv = af[r] + xw[H_ + col];
    float gv = ag[r] + xw[2 * H_ + col];
    float ov = ao[r] + xw[3 * H_ + col];
    float cn = sigmoidf_(fv) * c[m * H_ + col] + sigmoidf_(iv) * tanhf(gv);
    c[m * H_ + col] = cn;
    float hn = sigmoidf_(ov) * tanhf(cn);
    h_out[m * H_ + col] = (__bf16)hn;
    if (hs) hs[(size_t)(m * steps + t) * H_ + col] = (__bf16)hn;   // row = b*63+t
  }
}

// ---------------------------------------------------------------------------
// logits[1008, 32000] = hs @ fc_W^T + fc_b.
// Block = 8 waves = 8 M-tiles (128 rows) x one 32-wide N-pair.
// Per K-step:
//   - each wave TDM-DMAs its private 16x32 bf16 A-tile (hs rows) into LDS
//     (tensor_load_to_lds, tracked by TENSORcnt; wave-private => only
//     s_wait_tensorcnt needed, no cross-wave barrier for A)
//   - the block cooperatively stages the 32x32 fc_W tile, converting
//     f32->bf16 ONCE per block (8x fewer cvts than per-wave conversion)
// fc_W (131 MB f32) streams HBM->L2 once and is re-read from L2 by the 8
// M-groups; logits (129 MB f32) stream straight out: phase is HBM-bound.
// Dynamic LDS so TDM lds_addr offsets are 0-based: sA @ 0 (8KB), sB @ 8192.
// ---------------------------------------------------------------------------
__global__ void __launch_bounds__(256)
k_fc_lds(const __bf16* __restrict__ A, const float* __restrict__ Wf,
         const float* __restrict__ bias, float* __restrict__ out, int M) {
  extern __shared__ __align__(128) char smem[];
  __bf16* sA = (__bf16*)smem;                 // 8 waves * 16x32 bf16 = 8 KB
  __bf16* sB = (__bf16*)(smem + 8192);        // 32x32 bf16 = 2 KB
  const int lane = threadIdx.x & 31;
  const int wv   = threadIdx.x >> 5;          // 0..7 -> M-tile within group
  const unsigned wvu = __builtin_amdgcn_readfirstlane(wv);  // SGPR copy for D#
  const int mg   = blockIdx.y;                // M-group (0..7)
  const int n0   = blockIdx.x * 32;           // N-pair base
  const int m0   = mg * 128 + wv * 16;
  const __bf16* arow = A + (size_t)(mg * 128 + wvu * 16) * H_;  // uniform base
  const unsigned lds_off = wvu * (16 * 32 * 2);                 // bytes

  f32x8 acc0 = {}, acc1 = {};
  for (int k0 = 0; k0 < H_; k0 += 32) {
    __syncthreads();                          // all waves done reading sB/sA
    // A: wave-private 16x32 tile via Tensor Data Mover (async DMA -> LDS).
    tdm_load_tile_bf16(arow + k0, lds_off, 16u, (unsigned)H_);
    {  // B: 1024 f32 -> bf16, 4 contiguous per thread, cooperative
      const int idx = threadIdx.x * 4;
      const int r = idx >> 5, kk = idx & 31;
      const float* src = Wf + (size_t)(n0 + r) * H_ + k0 + kk;
#pragma unroll
      for (int j = 0; j < 4; ++j) sB[idx + j] = (__bf16)src[j];
    }
    if (k0 + 32 < H_) {
      __builtin_prefetch(Wf + (size_t)n0 * H_ + k0 + 32, 0, 3);
    }
    __builtin_amdgcn_s_wait_tensorcnt(0);     // A-tile landed in LDS
    __syncthreads();                          // sB visible to all waves
    bf16x16 a  = frag_ld_bf16(sA + wvu * (16 * 32), 32, 0, 0, lane);
    bf16x16 b0 = frag_ld_bf16(sB, 32, 0,  0, lane);
    bf16x16 b1 = frag_ld_bf16(sB, 32, 16, 0, lane);
    acc0 = wmma_bf16(a, b0, acc0);
    acc1 = wmma_bf16(a, b1, acc1);
  }
  if (m0 >= M) return;
  const int nidx = lane & 15, hf = lane >> 4;
#pragma unroll
  for (int r = 0; r < 8; ++r) {
    int m = hf * 8 + r;
    float* row = out + (size_t)(m0 + m) * V_;
    row[n0 + nidx]      = acc0[r] + bias[n0 + nidx];
    row[n0 + 16 + nidx] = acc1[r] + bias[n0 + 16 + nidx];
  }
}

// ---------------------------------------------------------------------------
// Host orchestration
// ---------------------------------------------------------------------------
static inline int cdiv(int a, int b) { return (a + b - 1) / b; }

extern "C" void kernel_launch(void* const* d_in, const int* in_sizes, int n_in,
                              void* d_out, int out_size, void* d_ws, size_t ws_size,
                              hipStream_t stream) {
  (void)in_sizes; (void)n_in; (void)out_size;

  const int*   src     = (const int*)  d_in[0];
  const int*   tgt     = (const int*)  d_in[1];
  const float* enc_emb = (const float*)d_in[2];
  const float* enc_Wih = (const float*)d_in[3];
  const float* enc_Whh = (const float*)d_in[4];
  const float* enc_bih = (const float*)d_in[5];
  const float* enc_bhh = (const float*)d_in[6];
  const float* dec_emb = (const float*)d_in[7];
  const float* dec_Wih = (const float*)d_in[8];
  const float* dec_Whh = (const float*)d_in[9];
  const float* dec_bih = (const float*)d_in[10];
  const float* dec_bhh = (const float*)d_in[11];
  const float* fc_W    = (const float*)d_in[12];
  const float* fc_b    = (const float*)d_in[13];
  float* out = (float*)d_out;

  // Workspace carve-up (all offsets 256B aligned; total ~81 MB).
  char*  ws  = (char*)d_ws;
  size_t off = 0;
  auto carve = [&](size_t bytes) -> char* {
    char* p = ws + off;
    off = (off + bytes + 255) & ~(size_t)255;
    return p;
  };
  __bf16* encX   = (__bf16*)carve((size_t)B_ * S_ * E_ * 2);       //  2.0 MB
  __bf16* decX   = (__bf16*)carve((size_t)B_ * TD * E_ * 2);       //  1.0 MB
  __bf16* eWih   = (__bf16*)carve((size_t)G4 * E_ * 2);            //  4.0 MB
  __bf16* eWhh   = (__bf16*)carve((size_t)G4 * H_ * 2);            //  8.0 MB
  __bf16* dWih   = (__bf16*)carve((size_t)G4 * E_ * 2);            //  4.0 MB
  __bf16* dWhh   = (__bf16*)carve((size_t)G4 * H_ * 2);            //  8.0 MB
  float*  biasE  = (float*) carve((size_t)G4 * 4);
  float*  biasD  = (float*) carve((size_t)G4 * 4);
  float*  XWe    = (float*) carve((size_t)B_ * S_ * G4 * 4);       // 32.0 MB
  float*  XWd    = (float*) carve((size_t)B_ * TD * G4 * 4);       // 15.8 MB
  __bf16* hbuf0  = (__bf16*)carve((size_t)B_ * H_ * 2);
  __bf16* hbuf1  = (__bf16*)carve((size_t)B_ * H_ * 2);
  float*  cbuf   = (float*) carve((size_t)B_ * H_ * 4);
  __bf16* hs_dec = (__bf16*)carve((size_t)M_FC_PAD * H_ * 2);      //  2.0 MB (padded)
  if (off > ws_size) return;  // workspace too small; bail safely

  const int TPB = 256;

  // 1) Weight casts f32 -> bf16 (one-time; recurrent weights then live in L2).
  k_cast_bf16<<<cdiv(G4 * E_, TPB), TPB, 0, stream>>>(enc_Wih, eWih, G4 * E_);
  k_cast_bf16<<<cdiv(G4 * H_, TPB), TPB, 0, stream>>>(enc_Whh, eWhh, G4 * H_);
  k_cast_bf16<<<cdiv(G4 * E_, TPB), TPB, 0, stream>>>(dec_Wih, dWih, G4 * E_);
  k_cast_bf16<<<cdiv(G4 * H_, TPB), TPB, 0, stream>>>(dec_Whh, dWhh, G4 * H_);
  k_bias_combine<<<cdiv(G4, TPB), TPB, 0, stream>>>(enc_bih, enc_bhh, biasE, G4);
  k_bias_combine<<<cdiv(G4, TPB), TPB, 0, stream>>>(dec_bih, dec_bhh, biasD, G4);

  // 2) Embedding gathers (decoder uses tgt[:, :-1]).
  k_embed<<<cdiv(B_ * S_ * E_, TPB), TPB, 0, stream>>>(src, enc_emb, encX, B_ * S_, S_, S_);
  k_embed<<<cdiv(B_ * TD * E_, TPB), TPB, 0, stream>>>(tgt, dec_emb, decX, B_ * TD, TD, T_);

  // 3) Hoisted input projections: XW = X @ Wih^T + (bih + bhh).
  {
    int waves = (B_ * S_ / 16) * (G4 / 32);
    k_gemm_xw<<<cdiv(waves, 8), TPB, 0, stream>>>(encX, eWih, biasE, XWe, B_ * S_, E_);
  }
  {
    int waves = (B_ * TD / 16) * (G4 / 32);
    k_gemm_xw<<<cdiv(waves, 8), TPB, 0, stream>>>(decX, dWih, biasD, XWd, B_ * TD, E_);
  }

  // 4) Recurrence. h0 = c0 = 0; stream-ordered step kernels, h ping-ponged.
  k_zero_state<<<cdiv(B_ * H_, TPB), TPB, 0, stream>>>(hbuf0, cbuf);
  __bf16* hb[2] = {hbuf0, hbuf1};
  int ping = 0;
  for (int t = 0; t < S_; ++t) {                 // encoder: 128 steps
    k_lstm_step<<<8, TPB, 0, stream>>>(hb[ping], hb[ping ^ 1], cbuf, eWhh,
                                       XWe, S_, t, (__bf16*)nullptr);
    ping ^= 1;
  }
  for (int t = 0; t < TD; ++t) {                 // decoder: 63 steps (h,c carry over)
    k_lstm_step<<<8, TPB, 0, stream>>>(hb[ping], hb[ping ^ 1], cbuf, dWhh,
                                       XWd, TD, t, hs_dec);
    ping ^= 1;
  }

  // 5) Vocabulary projection: TDM-fed, LDS-staged WMMA (HBM-bound phase).
  {
    dim3 grid(V_ / 32, M_FC_PAD / 128);          // 1000 x 8 blocks
    k_fc_lds<<<grid, TPB, 10240, stream>>>(hs_dec, fc_W, fc_b, out, M_FC);
  }
}
